// GIN_52218212385530
// MI455X (gfx1250) — compile-verified, matrix operand order checked by reading, not run on hardware
//
#include <hip/hip_runtime.h>
#include <hip/hip_bf16.h>

// ---------------- problem constants (from reference) ----------------
#define N_NODES   50000
#define N_EDGES   800000
#define DIM_IN    64
#define DIM_OUT   128
#define DIM_ED    16
#define NL_CONV   4
#define DIM_CLS   256
#define N_MOLS    2000
#define BN_INV_F  0.99999500003749980f   // 1/sqrt(1 + 1e-5)

typedef __attribute__((ext_vector_type(16))) __bf16 v16bf;
typedef __attribute__((ext_vector_type(8)))  float  v8f;

union BF16x16 {           // one 16-element bf16 fragment = 32 bytes = 2 x b128
  v16bf v;
  uint4 q[2];
};

// native f32 -> bf16 (v_cvt_pk_bf16_f32 pairs on gfx1250)
__device__ __forceinline__ __bf16 to_bf(float f) { return (__bf16)f; }

// ISA 7.12.2: 16-bit A/B fragment K index for element e (0..15) given lane half
__device__ __forceinline__ int frag_k(int e, int half) {
  int v = e >> 1, pos = e & 1;
  return (v < 4) ? (half * 8 + v * 2 + pos)
                 : (16 + half * 8 + (v - 4) * 2 + pos);
}

__device__ __forceinline__ float leaky1(float v) { return v > 0.0f ? v : 0.01f * v; }

// A fragment from row-major f32: lane's 16 K-values are two contiguous 8-float
// runs at p (= row + k0 + half*8) and p+16.  4 x global_load_b128 + 8 cvt_pk.
__device__ __forceinline__ v16bf make_a_frag(const float* __restrict__ p) {
  const float4 f0 = *(const float4*)(p + 0);
  const float4 f1 = *(const float4*)(p + 4);
  const float4 f2 = *(const float4*)(p + 16);
  const float4 f3 = *(const float4*)(p + 20);
  v16bf a;
  a[0]  = to_bf(f0.x); a[1]  = to_bf(f0.y); a[2]  = to_bf(f0.z); a[3]  = to_bf(f0.w);
  a[4]  = to_bf(f1.x); a[5]  = to_bf(f1.y); a[6]  = to_bf(f1.z); a[7]  = to_bf(f1.w);
  a[8]  = to_bf(f2.x); a[9]  = to_bf(f2.y); a[10] = to_bf(f2.z); a[11] = to_bf(f2.w);
  a[12] = to_bf(f3.x); a[13] = to_bf(f3.y); a[14] = to_bf(f3.z); a[15] = to_bf(f3.w);
  return a;
}

// A fragment with K=16 zero-padded to 32 (edge_attr): elems 0..7 from p, 8..15 = 0
__device__ __forceinline__ v16bf make_a_frag_k16(const float* __restrict__ p) {
  const float4 f0 = *(const float4*)(p + 0);
  const float4 f1 = *(const float4*)(p + 4);
  const __bf16 z = to_bf(0.0f);
  v16bf a;
  a[0] = to_bf(f0.x); a[1] = to_bf(f0.y); a[2] = to_bf(f0.z); a[3] = to_bf(f0.w);
  a[4] = to_bf(f1.x); a[5] = to_bf(f1.y); a[6] = to_bf(f1.z); a[7] = to_bf(f1.w);
  a[8] = z; a[9] = z; a[10] = z; a[11] = z; a[12] = z; a[13] = z; a[14] = z; a[15] = z;
  return a;
}

// ============================================================================
// Weight pre-pack: row-major f32 [Ksrc, C] -> bf16 WMMA B-fragment order.
// Tile (kt, nt): per-lane 16 bf16 stored contiguously (32B) -> 2 uint4.
// Rows >= Ksrc are zero (K padding for the ED=16 edge matmul).
// ============================================================================
__global__ void __launch_bounds__(256)
pack_w_kernel(const float* __restrict__ W, uint4* __restrict__ dst,
              int Ksrc, int C, int ktiles) {
  const int lane = threadIdx.x & 31;
  const int wave = blockIdx.x * (blockDim.x >> 5) + (threadIdx.x >> 5);
  const int tilesN = C >> 4;
  if (wave >= ktiles * tilesN) return;
  const int kt = wave / tilesN, nt = wave % tilesN;
  const int half = lane >> 4, nloc = lane & 15;
  BF16x16 b;
#pragma unroll
  for (int e = 0; e < 16; ++e) {
    const int k = kt * 32 + frag_k(e, half);
    const float v = (k < Ksrc) ? W[k * C + nt * 16 + nloc] : 0.0f;
    b.v[e] = to_bf(v);
  }
  uint4* q = dst + (wave * 32 + lane) * 2;
  q[0] = b.q[0];
  q[1] = b.q[1];
}

// ============================================================================
// Fused GINE message kernel: one wave = 16 edges.
//   e   = edge_attr[16x16] @ eW[16xC]   (bf16 WMMA, K padded 16->32)
//   msg = relu(h[src] + e + eb)
//   agg[dst] += msg                      (global_atomic_add_f32; agg pre-set to h)
// All gathers/scatters use uniform-base + i32 index (GVS saddr + scale_offset).
// ============================================================================
template <int C>
__global__ void __launch_bounds__(256)
gine_msg_kernel(const float* __restrict__ edge_attr,
                const uint4* __restrict__ eWpk, const float* __restrict__ eb,
                const float* __restrict__ hin,
                const int* __restrict__ src, const int* __restrict__ dst,
                float* __restrict__ agg) {
  constexpr int TILESN = C >> 4;
  const int lane = threadIdx.x & 31;
  const int wave = blockIdx.x * (blockDim.x >> 5) + (threadIdx.x >> 5);
  if (wave >= (N_EDGES >> 4)) return;   // wave-uniform: EXEC stays all-ones
  const int e0   = wave << 4;
  const int half = lane >> 4;
  const int nloc = lane & 15;

  const v16bf a = make_a_frag_k16(edge_attr + (size_t)(e0 + nloc) * DIM_ED + half * 8);
  if (e0 + 16 < N_EDGES)
    __builtin_prefetch(edge_attr + (size_t)(e0 + 16) * DIM_ED, 0, 1);

  // hoist i32 gather/scatter indices once per wave (n-tile offsets = immediates)
  int sIdx[8], dIdx[8];
#pragma unroll
  for (int r = 0; r < 8; ++r) {
    const int m = e0 + r + half * 8;     // C-layout: M = r + 8*(lane>=16)
    sIdx[r] = src[m] * C + nloc;
    dIdx[r] = dst[m] * C + nloc;
  }
  const int bqi = lane * 2;              // uint4 index into packed eW

#pragma unroll
  for (int nt = 0; nt < TILESN; ++nt) {
    BF16x16 b;
    b.q[0] = eWpk[bqi + nt * 64 + 0];
    b.q[1] = eWpk[bqi + nt * 64 + 1];
    v8f acc = {};
    acc = __builtin_amdgcn_wmma_f32_16x16x32_bf16(false, a, false, b.v,
                                                  (short)0, acc, false, false);
    const float ebn = eb[nloc + nt * 16];
#pragma unroll
    for (int r = 0; r < 8; ++r) {
      float v = acc[r] + ebn + hin[sIdx[r] + nt * 16];
      v = v > 0.0f ? v : 0.0f;           // relu
      atomicAdd(&agg[dIdx[r] + nt * 16], v);
    }
  }
}

// ============================================================================
// Generic multi-part WMMA GEMM + bias (+BN eval) (+0/1/2x LeakyReLU).
//   out[N,C] = act( gamma*((sum_p A_p @ W[pK..,:]) + bias)*BN_INV + beta )
// One wave computes a 16 x (NT*16) tile: A fragment loaded once per k-step and
// reused across NT wmma ops; all fragment addresses are immediate offsets.
// ============================================================================
template <int NPARTS, int PARTK, int C, int NT, int ACT, bool BN>
__global__ void __launch_bounds__(256)
wmma_gemm_kernel(const float* __restrict__ A0, const float* __restrict__ A1,
                 const float* __restrict__ A2, const float* __restrict__ A3,
                 const float* __restrict__ A4,
                 const uint4* __restrict__ Wpk, const float* __restrict__ bias,
                 const float* __restrict__ gamma, const float* __restrict__ beta,
                 float* __restrict__ out, int nRows) {
  constexpr int TILESN = C >> 4;
  constexpr int NBLK   = TILESN / NT;
  const float* parts[5] = {A0, A1, A2, A3, A4};
  const int lane = threadIdx.x & 31;
  const int wave = blockIdx.x * (blockDim.x >> 5) + (threadIdx.x >> 5);
  if (wave >= (nRows >> 4) * NBLK) return;      // wave-uniform
  const int m0   = (wave / NBLK) << 4;
  const int nb   = (wave % NBLK) * NT;          // first n-tile of this wave
  const int half = lane >> 4;
  const int nloc = lane & 15;

  const int bqi = (nb * 32 + lane) * 2;         // uint4 index; k/t offsets immediate

  v8f acc[NT] = {};
#pragma unroll
  for (int p = 0; p < NPARTS; ++p) {
    const float* Ap = parts[p] + (size_t)(m0 + nloc) * PARTK + half * 8;
#pragma unroll
    for (int k0 = 0; k0 < PARTK; k0 += 32) {
      const v16bf a = make_a_frag(Ap + k0);
      const int gkt = (p * PARTK + k0) >> 5;
#pragma unroll
      for (int t = 0; t < NT; ++t) {
        BF16x16 b;
        b.q[0] = Wpk[bqi + (gkt * TILESN + t) * 64 + 0];
        b.q[1] = Wpk[bqi + (gkt * TILESN + t) * 64 + 1];
        acc[t] = __builtin_amdgcn_wmma_f32_16x16x32_bf16(false, a, false, b.v,
                                                         (short)0, acc[t], false, false);
      }
    }
  }

  const int nBase = nb * 16 + nloc;
  const int oBase = (m0 + half * 8) * C + nBase;     // i32 store index (GVS)
#pragma unroll
  for (int t = 0; t < NT; ++t) {
    const float bi = bias[nBase + t * 16];
    float g = 1.0f, be = 0.0f;
    if (BN) { g = gamma[nBase + t * 16]; be = beta[nBase + t * 16]; }
#pragma unroll
    for (int r = 0; r < 8; ++r) {
      float v = acc[t][r] + bi;
      if (BN) v = g * (v * BN_INV_F) + be;
      if (ACT >= 1) v = leaky1(v);      // leaky inside gine nn
      if (ACT >= 2) v = leaky1(v);      // outer leaky in forward()
      out[oBase + r * C + t * 16] = v;
    }
  }
}

// ---------------- elementwise helpers (float4-vectorized) ----------------
__global__ void copy4_kernel(const float4* __restrict__ s, float4* __restrict__ d, int n4) {
  int i = blockIdx.x * blockDim.x + threadIdx.x;
  if (i < n4) d[i] = s[i];
}
__global__ void zero4_kernel(float4* __restrict__ d, int n4) {
  int i = blockIdx.x * blockDim.x + threadIdx.x;
  if (i < n4) d[i] = make_float4(0.f, 0.f, 0.f, 0.f);
}
__global__ void pool_scatter_kernel(const float4* __restrict__ h, const int* __restrict__ batch,
                                    float* __restrict__ pool, int n4) {
  int i = blockIdx.x * blockDim.x + threadIdx.x;     // i over N * (128/4)
  if (i >= n4) return;
  const int node = i >> 5, c4 = i & 31;
  const float4 v = h[i];
  const int p = batch[node] * DIM_OUT + c4 * 4;
  atomicAdd(&pool[p + 0], v.x); atomicAdd(&pool[p + 1], v.y);
  atomicAdd(&pool[p + 2], v.z); atomicAdd(&pool[p + 3], v.w);
}
__global__ void pool_gather_kernel(const float4* __restrict__ pool, const int* __restrict__ batch,
                                   float4* __restrict__ hp, int n4) {
  int i = blockIdx.x * blockDim.x + threadIdx.x;
  if (i >= n4) return;
  const int node = i >> 5, c4 = i & 31;
  hp[i] = pool[batch[node] * (DIM_OUT / 4) + c4];
}
__global__ void final_kernel(const float* __restrict__ h, const float* __restrict__ fW,
                             const float* __restrict__ fb, float* __restrict__ out, int n) {
  int i = blockIdx.x * blockDim.x + threadIdx.x;
  if (i >= n) return;
  const float4* row = (const float4*)(h + (size_t)i * DIM_CLS);
  const float4* w4  = (const float4*)fW;
  float s = fb[0];
#pragma unroll 4
  for (int k = 0; k < DIM_CLS / 4; ++k) {
    const float4 r = row[k], w = w4[k];
    s = fmaf(r.x, w.x, s); s = fmaf(r.y, w.y, s);
    s = fmaf(r.z, w.z, s); s = fmaf(r.w, w.w, s);
  }
  out[i] = 1.0f / (1.0f + __expf(-s));
}

// ============================================================================
extern "C" void kernel_launch(void* const* d_in, const int* in_sizes, int n_in,
                              void* d_out, int out_size, void* d_ws, size_t ws_size,
                              hipStream_t stream) {
  (void)in_sizes; (void)n_in; (void)out_size; (void)ws_size;
  const float* x     = (const float*)d_in[0];
  const int*   eidx  = (const int*)  d_in[1];
  const float* eattr = (const float*)d_in[2];
  const int*   batch = (const int*)  d_in[3];
  const float* c1W   = (const float*)d_in[4];
  const float* c1b   = (const float*)d_in[5];
  const float* c1g   = (const float*)d_in[6];
  const float* c1be  = (const float*)d_in[7];
  const float* c1eW  = (const float*)d_in[8];
  const float* c1eb  = (const float*)d_in[9];
  const float* cW    = (const float*)d_in[10];
  const float* cb    = (const float*)d_in[11];
  const float* cg    = (const float*)d_in[12];
  const float* cbe   = (const float*)d_in[13];
  const float* ceW   = (const float*)d_in[14];
  const float* ceb   = (const float*)d_in[15];
  const float* cls1W = (const float*)d_in[16];
  const float* cls1b = (const float*)d_in[17];
  const float* clsW  = (const float*)d_in[18];
  const float* clsb  = (const float*)d_in[19];
  const float* fW    = (const float*)d_in[20];
  const float* fb    = (const float*)d_in[21];
  const int* src = eidx;
  const int* dst = eidx + N_EDGES;

  // ---- workspace: f32 buffers then packed-bf16 weight area (~284 MB total)
  float* ws = (float*)d_ws;
  size_t off = 0;
  float* agg  = ws + off; off += (size_t)N_NODES * DIM_OUT;
  float* h0   = ws + off; off += (size_t)N_NODES * DIM_OUT;
  float* hs[NL_CONV];
  for (int i = 0; i < NL_CONV; ++i) { hs[i] = ws + off; off += (size_t)N_NODES * DIM_OUT; }
  float* hp   = ws + off; off += (size_t)N_NODES * DIM_OUT;
  float* pool = ws + off; off += (size_t)N_MOLS  * DIM_OUT;
  float* cb1  = ws + off; off += (size_t)N_NODES * DIM_CLS;
  float* cb2  = ws + off; off += (size_t)N_NODES * DIM_CLS;

  uint4* wpk = (uint4*)(ws + off);      // off is a multiple of 4 floats -> 16B ok
  size_t qo = 0;                        // offsets in uint4 units (tile = 64 uint4)
  uint4* pk_c1eW = wpk + qo; qo += (size_t)4  * 64;                  // 1kt x 4nt
  uint4* pk_ceW[NL_CONV];
  for (int i = 0; i < NL_CONV; ++i) { pk_ceW[i] = wpk + qo; qo += (size_t)8 * 64; }
  uint4* pk_c1W  = wpk + qo; qo += (size_t)16 * 64;                  // 2kt x 8nt
  uint4* pk_cW[NL_CONV];
  for (int i = 0; i < NL_CONV; ++i) { pk_cW[i] = wpk + qo; qo += (size_t)32 * 64; }
  uint4* pk_cls1 = wpk + qo; qo += (size_t)320 * 64;                 // 20kt x 16nt
  uint4* pk_cls[2];
  for (int i = 0; i < 2; ++i) { pk_cls[i] = wpk + qo; qo += (size_t)128 * 64; }

  const dim3 blk(256);
  auto ewb = [](int n) { return (n + 255) / 256; };     // elementwise blocks
  auto pkb = [](int tiles) { return (tiles + 7) / 8; }; // 8 waves/block
  auto gwb = [](int waves) { return (waves + 7) / 8; };
  const int msgBlocks  = gwb(N_EDGES / 16);
  const int MT = N_NODES / 16;                          // 3125 row tiles
  const int gemmBlkConv = gwb(MT * 2);                  // C=128, NT=4 -> NBLK=2
  const int gemmBlkCls  = gwb(MT * 4);                  // C=256, NT=4 -> NBLK=4

  // ---- pack all weights into WMMA fragment order (bf16), ~740 KB
  pack_w_kernel<<<pkb(4),   blk, 0, stream>>>(c1eW, pk_c1eW, DIM_ED, DIM_IN,  1);
  for (int i = 0; i < NL_CONV; ++i)
    pack_w_kernel<<<pkb(8), blk, 0, stream>>>(ceW + (size_t)i * DIM_ED * DIM_OUT,
                                              pk_ceW[i], DIM_ED, DIM_OUT, 1);
  pack_w_kernel<<<pkb(16),  blk, 0, stream>>>(c1W, pk_c1W, DIM_IN, DIM_OUT, 2);
  for (int i = 0; i < NL_CONV; ++i)
    pack_w_kernel<<<pkb(32), blk, 0, stream>>>(cW + (size_t)i * DIM_OUT * DIM_OUT,
                                               pk_cW[i], DIM_OUT, DIM_OUT, 4);
  pack_w_kernel<<<pkb(320), blk, 0, stream>>>(cls1W, pk_cls1, DIM_OUT * 5, DIM_CLS, 20);
  for (int i = 0; i < 2; ++i)
    pack_w_kernel<<<pkb(128), blk, 0, stream>>>(clsW + (size_t)i * DIM_CLS * DIM_CLS,
                                                pk_cls[i], DIM_CLS, DIM_CLS, 8);

  // ---- conv1: agg = x; agg += relu(x[src] + eattr@eW + eb); h0 = BN/2xleaky(agg@W+b)
  copy4_kernel<<<ewb(N_NODES * DIM_IN / 4), blk, 0, stream>>>(
      (const float4*)x, (float4*)agg, N_NODES * DIM_IN / 4);
  gine_msg_kernel<DIM_IN><<<msgBlocks, blk, 0, stream>>>(
      eattr, pk_c1eW, c1eb, x, src, dst, agg);
  wmma_gemm_kernel<1, DIM_IN, DIM_OUT, 4, 2, true><<<gemmBlkConv, blk, 0, stream>>>(
      agg, nullptr, nullptr, nullptr, nullptr, pk_c1W, c1b, c1g, c1be, h0, N_NODES);

  // ---- conv layers 0..3
  const float* hin = h0;
  for (int i = 0; i < NL_CONV; ++i) {
    copy4_kernel<<<ewb(N_NODES * DIM_OUT / 4), blk, 0, stream>>>(
        (const float4*)hin, (float4*)agg, N_NODES * DIM_OUT / 4);
    gine_msg_kernel<DIM_OUT><<<msgBlocks, blk, 0, stream>>>(
        eattr, pk_ceW[i], ceb + (size_t)i * DIM_OUT, hin, src, dst, agg);
    wmma_gemm_kernel<1, DIM_OUT, DIM_OUT, 4, 2, true><<<gemmBlkConv, blk, 0, stream>>>(
        agg, nullptr, nullptr, nullptr, nullptr, pk_cW[i],
        cb + (size_t)i * DIM_OUT, cg + (size_t)i * DIM_OUT, cbe + (size_t)i * DIM_OUT,
        hs[i], N_NODES);
    hin = hs[i];
  }

  // ---- global_add_pool over batch, then broadcast back to nodes
  zero4_kernel<<<ewb(N_MOLS * DIM_OUT / 4), blk, 0, stream>>>(
      (float4*)pool, N_MOLS * DIM_OUT / 4);
  pool_scatter_kernel<<<ewb(N_NODES * DIM_OUT / 4), blk, 0, stream>>>(
      (const float4*)hs[NL_CONV - 1], batch, pool, N_NODES * DIM_OUT / 4);
  pool_gather_kernel<<<ewb(N_NODES * DIM_OUT / 4), blk, 0, stream>>>(
      (const float4*)pool, batch, (float4*)hp, N_NODES * DIM_OUT / 4);

  // ---- classifier1: concat(hs0..hs3, hp) @ cls1_W + b   (5-part K=128, no act)
  wmma_gemm_kernel<5, DIM_OUT, DIM_CLS, 4, 0, false><<<gemmBlkCls, blk, 0, stream>>>(
      hs[0], hs[1], hs[2], hs[3], hp, pk_cls1, cls1b, nullptr, nullptr, cb1, N_NODES);

  // ---- classifier hidden layers: leaky(h @ W + b)
  wmma_gemm_kernel<1, DIM_CLS, DIM_CLS, 4, 1, false><<<gemmBlkCls, blk, 0, stream>>>(
      cb1, nullptr, nullptr, nullptr, nullptr, pk_cls[0], clsb, nullptr, nullptr,
      cb2, N_NODES);
  wmma_gemm_kernel<1, DIM_CLS, DIM_CLS, 4, 1, false><<<gemmBlkCls, blk, 0, stream>>>(
      cb2, nullptr, nullptr, nullptr, nullptr, pk_cls[1], clsb + DIM_CLS, nullptr, nullptr,
      cb1, N_NODES);

  // ---- final: sigmoid(h @ final_W + final_b)
  final_kernel<<<ewb(N_NODES), blk, 0, stream>>>(cb1, fW, fb, (float*)d_out, N_NODES);
}
